// GNN_26285199851684
// MI455X (gfx1250) — matrix operand order repaired and checked by reading, not compiled, not run on hardware
//
#include <hip/hip_runtime.h>
#include <math.h>

#define N_NODES 50000
#define N_EDGES 800000
#define NG      8
#define XSTRIDE 132        // FEAT + SUB
#define BN_EPS  1e-5f

typedef __attribute__((ext_vector_type(16))) __bf16 v16bf;
typedef __attribute__((ext_vector_type(8)))  __bf16 v8bf;
typedef __attribute__((ext_vector_type(8)))  float  v8f;

// ---------------- small helpers ----------------
__device__ __forceinline__ unsigned short f2bf(float f){
    unsigned u = __float_as_uint(f);
    u += 0x7FFFu + ((u >> 16) & 1u);          // RNE
    return (unsigned short)(u >> 16);
}
__device__ __forceinline__ float bf2f(unsigned short b){
    return __uint_as_float(((unsigned)b) << 16);
}
__device__ __forceinline__ unsigned pack2(float a, float b){
    return (unsigned)f2bf(a) | ((unsigned)f2bf(b) << 16);
}
// order-preserving encode for signed float max via uint atomicMax
__device__ __forceinline__ unsigned encf(float f){
    unsigned u = __float_as_uint(f);
    return u ^ (((unsigned)((int)u >> 31)) | 0x80000000u);
}
__device__ __forceinline__ float decf(unsigned e){
    unsigned u = (e & 0x80000000u) ? (e ^ 0x80000000u) : ~e;
    return __uint_as_float(u);
}
__device__ __forceinline__ bool finitef(float v){
    return (__float_as_uint(v) & 0x7F800000u) != 0x7F800000u;
}
__device__ __forceinline__ float sigf(float x){ return 1.f / (1.f + expf(-x)); }

// ---------------- WMMA helpers (CDNA5 16x16x32 bf16) ----------------
__device__ __forceinline__ v8f wmma_bf(v16bf a, v16bf b, v8f c){
    return __builtin_amdgcn_wmma_f32_16x16x32_bf16(false, a, false, b, (short)0, c, false, false);
}
// A operand: lane holds row m=lane%16; elems[0..7]=K(kt*32 + half*8 ..),
// elems[8..15]=K(kt*32+16+half*8 ..)  (two 16B chunks -> ds_load_b128 x2)
__device__ __forceinline__ v16bf ld_a(const __bf16* row, int kt, int half){
    union { v16bf v; v8bf h[2]; } u;
    u.h[0] = *(const v8bf*)(row + kt*32 + half*8);
    u.h[1] = *(const v8bf*)(row + kt*32 + 16 + half*8);
    return u.v;
}
// B operand: lane holds column n=lane%16; elems in same K order as ld_a (strided gather, row stride ld)
__device__ __forceinline__ v16bf ld_b(const __bf16* q, int ld){
    v16bf v;
    #pragma unroll
    for (int s = 0; s < 8; ++s){ v[s] = q[s*ld]; v[8+s] = q[(16+s)*ld]; }
    return v;
}

// ---------------- global scratch layout ----------------
struct Small {
    float cs1[128], csq1[128];      // layer1 BN raw sums
    float csC[128], csqC[128];      // conv BN raw sums
    float csS[128], csqS[128];      // score BN raw sums
    float subs1[4],  subM2[16];     // sub first/second moments
    float gcnt[8];
    float a1[128], b1[128];         // BN affine: y*a+b
    float aC[128], bC[128];
    float aE[128], bE[128];
    float aS[128], bS[128];
    float gsum[1024], gsq[1024];
    float gmean[1024], gstd[1024];
    float wsum[1024];
    float denom[8];
    unsigned out_enc[16];
    unsigned smax_enc[8];
};

struct P {
    const float *x; const int *ei; const int *batch;
    const float *attn_W1,*attn_b1,*attn_W2,*attn_b2;
    const float *first_W,*first_b,*first_g,*first_beta;
    const float *proj0_W,*proj0_b,*proj1_W,*proj1_b;
    const float *scale0,*scale1;
    const float *lin0_W,*lin0_b,*lin1_W,*lin1_b;
    const float *conv_W,*conv_b,*conv_g,*conv_beta;
    const float *enh_W,*enh_b,*enh_g,*enh_beta;
    const float *gate_W,*gate_b;
    const float *score_W1,*score_b1,*score_g,*score_beta;
    const float *score_W2,*score_b2,*mix;
    unsigned *h1b;          // h1 as packed bf16 (N x 128) -> 64 uints/row
    float *y1;              // N x 128 scratch (layer1 pre, reused for score pre)
    float *hfus;            // N x 128
    float *logit;           // N
    unsigned short *wbC,*wbG,*wbS;   // bf16 weights 256x128
    unsigned *hnew;         // N x 128 (uint-encoded nonneg float max)
    float *attn_sum; float *degree;
    Small *S;
    float *out,*Z,*H;
};

// ---------------- weight conversion ----------------
__global__ __launch_bounds__(256) void k_cvtW(P p){
    int i = blockIdx.x*256 + threadIdx.x;
    if (i < 256*128){
        p.wbC[i] = f2bf(p.conv_W[i]);
        p.wbG[i] = f2bf(p.gate_W[i]);
        p.wbS[i] = f2bf(p.score_W1[i]);
    }
}

// ---------------- edge attention MLP ----------------
__global__ __launch_bounds__(256) void k_edgeattn(P p){
    int e = blockIdx.x*256 + threadIdx.x;
    if (e >= N_EDGES) return;
    int s = p.ei[e], d = p.ei[N_EDGES + e];
    float pr[8];
    #pragma unroll
    for (int k = 0; k < 4; ++k) pr[k]   = p.x[(size_t)s*XSTRIDE + 128 + k];
    #pragma unroll
    for (int k = 0; k < 4; ++k) pr[4+k] = p.x[(size_t)d*XSTRIDE + 128 + k];
    float logit = p.attn_b2[0];
    #pragma unroll
    for (int j = 0; j < 16; ++j){
        float hm = p.attn_b1[j];
        #pragma unroll
        for (int i = 0; i < 8; ++i) hm += pr[i] * p.attn_W1[i*16 + j];
        hm = fmaxf(hm, 0.f);
        logit += hm * p.attn_W2[j];
    }
    float w = sigf(logit);
    atomicAdd(&p.attn_sum[d], w);
    atomicAdd(&p.degree[d], 1.0f);
}

// ---------------- sub moments + graph counts ----------------
__global__ __launch_bounds__(256) void k_submom(P p){
    __shared__ float acc[20];
    __shared__ float cnt[8];
    int tid = threadIdx.x;
    if (tid < 20) acc[tid] = 0.f;
    if (tid < 8)  cnt[tid] = 0.f;
    __syncthreads();
    int n = blockIdx.x*256 + tid;
    if (n < N_NODES){
        float s[4];
        #pragma unroll
        for (int k = 0; k < 4; ++k) s[k] = p.x[(size_t)n*XSTRIDE + 128 + k];
        #pragma unroll
        for (int k = 0; k < 4; ++k) atomicAdd(&acc[k], s[k]);
        #pragma unroll
        for (int k = 0; k < 4; ++k)
            #pragma unroll
            for (int l = 0; l < 4; ++l) atomicAdd(&acc[4 + k*4 + l], s[k]*s[l]);
        atomicAdd(&cnt[p.batch[n]], 1.0f);
    }
    __syncthreads();
    if (tid < 4)             atomicAdd(&p.S->subs1[tid], acc[tid]);
    if (tid >= 4 && tid < 20) atomicAdd(&p.S->subM2[tid-4], acc[tid]);
    if (tid >= 32 && tid < 40) atomicAdd(&p.S->gcnt[tid-32], cnt[tid-32]);
}

// ---------------- layer1 GEMM (feat @ first_W) + BN stats ----------------
__global__ __launch_bounds__(128) void k_layer1(P p){
    __shared__ __align__(16) __bf16 Wl[128*136];       // padded rows
    __shared__ __align__(16) __bf16 At[4][16*136];
    int tid = threadIdx.x, wv = tid >> 5, ln = tid & 31;
    for (int i = tid; i < 128*128; i += 128){
        int r = i >> 7, c = i & 127;
        ((unsigned short*)Wl)[r*136 + c] = f2bf(p.first_W[i]);
    }
    int rowbase = blockIdx.x*64 + wv*16;
    bool fullt = (rowbase + 16) <= N_NODES;
    {   // stage 16x128 A tile (2 lanes per row)
        int m = ln >> 1, half = ln & 1;
        int n = rowbase + m;
        unsigned short* arow = (unsigned short*)&At[wv][m*136];
        if (n < N_NODES){
            const float* xr = p.x + (size_t)n*XSTRIDE + half*64;
            #pragma unroll
            for (int j = 0; j < 64; j += 4){
                float4 v = *(const float4*)(xr + j);
                arow[half*64+j+0] = f2bf(v.x);
                arow[half*64+j+1] = f2bf(v.y);
                arow[half*64+j+2] = f2bf(v.z);
                arow[half*64+j+3] = f2bf(v.w);
            }
        } else {
            for (int j = 0; j < 64; ++j) arow[half*64+j] = 0;
        }
    }
    __syncthreads();
    const __bf16* arow = &At[wv][(ln & 15)*136];
    int half = ln >> 4;
    for (int nt = 0; nt < 8; ++nt){
        v8f acc = {0,0,0,0,0,0,0,0};
        #pragma unroll
        for (int kt = 0; kt < 4; ++kt){
            v16bf a = ld_a(arow, kt, half);
            v16bf b = ld_b(&Wl[(kt*32 + half*8)*136 + nt*16 + (ln & 15)], 136);
            acc = wmma_bf(a, b, acc);
        }
        int col = nt*16 + (ln & 15);
        float bias = p.first_b[col];
        float ps = 0.f, pq = 0.f;
        if (fullt){
            #pragma unroll
            for (int k = 0; k < 8; ++k){
                int r = rowbase + k + 8*half;
                float y = acc[k] + bias;
                p.y1[(size_t)r*128 + col] = y;
                ps += y; pq += y*y;
            }
        } else {
            #pragma unroll
            for (int k = 0; k < 8; ++k){
                int r = rowbase + k + 8*half;
                float y = acc[k] + bias;
                if (r < N_NODES){
                    p.y1[(size_t)r*128 + col] = y;
                    ps += y; pq += y*y;
                }
            }
        }
        ps += __shfl_xor(ps, 16, 32);
        pq += __shfl_xor(pq, 16, 32);
        if (ln < 16){
            atomicAdd(&p.S->cs1[col], ps);
            atomicAdd(&p.S->csq1[col], pq);
        }
    }
}

// ---------------- BN coefs: layer1 + enh (analytic from sub moments) ----------------
__global__ __launch_bounds__(128) void k_coef1(P p){
    int c = threadIdx.x;
    if (c >= 128) return;
    float invN = 1.0f / (float)N_NODES;
    float m = p.S->cs1[c]*invN;
    float v = p.S->csq1[c]*invN - m*m;
    float a = p.first_g[c] * rsqrtf(v + BN_EPS);
    p.S->a1[c] = a;
    p.S->b1[c] = p.first_beta[c] - m*a;

    float b = p.enh_b[c];
    float St = 0.f;
    #pragma unroll
    for (int k = 0; k < 4; ++k) St += p.S->subs1[k] * p.enh_W[k*128 + c];
    float mean = St*invN + b;
    float E2 = 0.f;
    #pragma unroll
    for (int k = 0; k < 4; ++k)
        #pragma unroll
        for (int l = 0; l < 4; ++l)
            E2 += p.S->subM2[k*4 + l] * p.enh_W[k*128 + c] * p.enh_W[l*128 + c];
    E2 = E2*invN + 2.f*b*(St*invN) + b*b;
    float var = E2 - mean*mean;
    float ae = p.enh_g[c] * rsqrtf(var + BN_EPS);
    p.S->aE[c] = ae;
    p.S->bE[c] = p.enh_beta[c] - mean*ae;
}

// ---------------- apply layer1 BN + relu + proj0; z0; out seg_max; h1 -> bf16 ----------------
__global__ __launch_bounds__(256) void k_apply1(P p){
    int wv = threadIdx.x >> 5, ln = threadIdx.x & 31;
    int n = blockIdx.x*8 + wv;
    if (n >= N_NODES) return;
    float sub[4];
    #pragma unroll
    for (int k = 0; k < 4; ++k) sub[k] = p.x[(size_t)n*XSTRIDE + 128 + k];
    float s0 = p.scale0[0];
    float pz0 = 0.f, pz1 = 0.f;
    #pragma unroll
    for (int j = 0; j < 4; ++j){
        int c = ln + 32*j;
        float y = p.y1[(size_t)n*128 + c];
        float h = fmaxf(p.S->a1[c]*y + p.S->b1[c], 0.f);
        float pj = p.proj0_b[c];
        #pragma unroll
        for (int k = 0; k < 4; ++k) pj += sub[k] * p.proj0_W[k*128 + c];
        h += s0*pj;
        ((unsigned short*)p.h1b)[(size_t)n*128 + c] = f2bf(h);
        pz0 += h * p.lin0_W[c*2 + 0];
        pz1 += h * p.lin0_W[c*2 + 1];
    }
    #pragma unroll
    for (int o = 16; o > 0; o >>= 1){
        pz0 += __shfl_xor(pz0, o, 32);
        pz1 += __shfl_xor(pz1, o, 32);
    }
    if (ln == 0){
        p.Z[(size_t)n*2 + 0] = pz0;
        p.Z[(size_t)n*2 + 1] = pz1;
        int g = p.batch[n];
        atomicMax(&p.S->out_enc[g*2 + 0], encf(pz0));
        atomicMax(&p.S->out_enc[g*2 + 1], encf(pz1));
    }
}

// ---------------- edge tile staging: msg_in = [x_i, x_j - x_i] in bf16 LDS ----------------
__device__ __forceinline__ unsigned diffpack(unsigned a, unsigned b){
    float lo = bf2f((unsigned short)(a & 0xFFFFu)) - bf2f((unsigned short)(b & 0xFFFFu));
    float hi = bf2f((unsigned short)(a >> 16))     - bf2f((unsigned short)(b >> 16));
    return pack2(lo, hi);
}
__device__ __forceinline__ void stage_edges(const P& p, int t, int ln,
                                            unsigned short* atile, int* dsts){
    int m = ln >> 1, half = ln & 1;
    int e = t*16 + m;
    int s = p.ei[e], d = p.ei[N_EDGES + e];
    unsigned short* arow = atile + m*264;
    const unsigned* hi = p.h1b + (size_t)d*64 + half*32;
    const unsigned* hj = p.h1b + (size_t)s*64 + half*32;
    unsigned* ai = (unsigned*)arow + half*32;        // x_i
    unsigned* ad = (unsigned*)arow + 64 + half*32;   // x_j - x_i
    #pragma unroll
    for (int j = 0; j < 32; j += 4){
        uint4 vi = *(const uint4*)(hi + j);
        uint4 vj = *(const uint4*)(hj + j);
        *(uint4*)(ai + j) = vi;
        uint4 dv;
        dv.x = diffpack(vj.x, vi.x);
        dv.y = diffpack(vj.y, vi.y);
        dv.z = diffpack(vj.z, vi.z);
        dv.w = diffpack(vj.w, vi.w);
        *(uint4*)(ad + j) = dv;
    }
    if (dsts && half == 0) dsts[m] = d;
}

// ---------------- conv pass 1: BN column stats of msg_pre ----------------
#define CONV_BLOCKS 640
__global__ __launch_bounds__(128) void k_conv_stats(P p){
    __shared__ __align__(16) __bf16 At[4][16*264];
    __shared__ float sS[128], sQ[128];
    int tid = threadIdx.x, wv = tid >> 5, ln = tid & 31;
    if (tid < 128){ sS[tid] = 0.f; sQ[tid] = 0.f; }
    __syncthreads();
    const __bf16* Wg = (const __bf16*)p.wbC;
    int half = ln >> 4;
    for (int t = blockIdx.x*4 + wv; t < N_EDGES/16; t += CONV_BLOCKS*4){
        stage_edges(p, t, ln, (unsigned short*)&At[wv][0], nullptr);
        const __bf16* arow = &At[wv][(ln & 15)*264];
        for (int nt = 0; nt < 8; ++nt){
            v8f acc = {0,0,0,0,0,0,0,0};
            #pragma unroll
            for (int kt = 0; kt < 8; ++kt){
                v16bf a = ld_a(arow, kt, half);
                v16bf b = ld_b(&Wg[(kt*32 + half*8)*128 + nt*16 + (ln & 15)], 128);
                acc = wmma_bf(a, b, acc);
            }
            int col = nt*16 + (ln & 15);
            float bias = p.conv_b[col];
            float ps = 0.f, pq = 0.f;
            #pragma unroll
            for (int k = 0; k < 8; ++k){
                float y = acc[k] + bias;
                ps += y; pq += y*y;
            }
            ps += __shfl_xor(ps, 16, 32);
            pq += __shfl_xor(pq, 16, 32);
            if (ln < 16){
                atomicAdd(&sS[col], ps);
                atomicAdd(&sQ[col], pq);
            }
        }
    }
    __syncthreads();
    if (tid < 128){
        atomicAdd(&p.S->csC[tid],  sS[tid]);
        atomicAdd(&p.S->csqC[tid], sQ[tid]);
    }
}

__global__ __launch_bounds__(128) void k_coefC(P p){
    int c = threadIdx.x;
    if (c >= 128) return;
    float inv = 1.0f / (float)N_EDGES;
    float m = p.S->csC[c]*inv;
    float v = p.S->csqC[c]*inv - m*m;
    float a = p.conv_g[c] * rsqrtf(v + BN_EPS);
    p.S->aC[c] = a;
    p.S->bC[c] = p.conv_beta[c] - m*a;
}

// ---------------- conv pass 2: recompute, BN+relu, seg_max into hnew ----------------
__global__ __launch_bounds__(128) void k_conv_apply(P p){
    __shared__ __align__(16) __bf16 At[4][16*264];
    __shared__ int dsts[4][16];
    int tid = threadIdx.x, wv = tid >> 5, ln = tid & 31;
    const __bf16* Wg = (const __bf16*)p.wbC;
    int half = ln >> 4;
    for (int t = blockIdx.x*4 + wv; t < N_EDGES/16; t += CONV_BLOCKS*4){
        stage_edges(p, t, ln, (unsigned short*)&At[wv][0], &dsts[wv][0]);
        int drow[8];
        #pragma unroll
        for (int k = 0; k < 8; ++k) drow[k] = dsts[wv][k + 8*half];
        const __bf16* arow = &At[wv][(ln & 15)*264];
        for (int nt = 0; nt < 8; ++nt){
            v8f acc = {0,0,0,0,0,0,0,0};
            #pragma unroll
            for (int kt = 0; kt < 8; ++kt){
                v16bf a = ld_a(arow, kt, half);
                v16bf b = ld_b(&Wg[(kt*32 + half*8)*128 + nt*16 + (ln & 15)], 128);
                acc = wmma_bf(a, b, acc);
            }
            int col = nt*16 + (ln & 15);
            float aC = p.S->aC[col], bC = p.S->bC[col], bias = p.conv_b[col];
            #pragma unroll
            for (int k = 0; k < 8; ++k){
                float msg = fmaxf(aC*(acc[k] + bias) + bC, 0.f);
                atomicMax(&p.hnew[(size_t)drow[k]*128 + col], __float_as_uint(msg));
            }
        }
    }
}

// ---------------- node update: h2 = hnew*attn + scale1*proj1 ; z1 ; graph sums ----------------
#define NB2 256
__global__ __launch_bounds__(256) void k_node2(P p){
    __shared__ float gS[1024], gQ[1024];
    int tid = threadIdx.x, wv = tid >> 5, ln = tid & 31;
    for (int i = tid; i < 1024; i += 256){ gS[i] = 0.f; gQ[i] = 0.f; }
    __syncthreads();
    float s1 = p.scale1[0];
    for (int n = blockIdx.x*8 + wv; n < N_NODES; n += NB2*8){
        float sub[4];
        #pragma unroll
        for (int k = 0; k < 4; ++k) sub[k] = p.x[(size_t)n*XSTRIDE + 128 + k];
        float att = p.attn_sum[n] / fmaxf(p.degree[n], 1.0f);
        int g = p.batch[n];
        float pz0 = 0.f, pz1 = 0.f;
        #pragma unroll
        for (int j = 0; j < 4; ++j){
            int c = ln + 32*j;
            float hn = __uint_as_float(p.hnew[(size_t)n*128 + c]);  // nonneg max
            float pj = p.proj1_b[c];
            #pragma unroll
            for (int k = 0; k < 4; ++k) pj += sub[k] * p.proj1_W[k*128 + c];
            float h = hn*att + s1*pj;
            p.H[(size_t)n*128 + c] = h;
            pz0 += h * p.lin1_W[c*2 + 0];
            pz1 += h * p.lin1_W[c*2 + 1];
            atomicAdd(&gS[g*128 + c], h);
            atomicAdd(&gQ[g*128 + c], h*h);
        }
        #pragma unroll
        for (int o = 16; o > 0; o >>= 1){
            pz0 += __shfl_xor(pz0, o, 32);
            pz1 += __shfl_xor(pz1, o, 32);
        }
        if (ln == 0){
            p.Z[(size_t)n*2 + 0] += pz0;
            p.Z[(size_t)n*2 + 1] += pz1;
        }
    }
    __syncthreads();
    for (int i = tid; i < 1024; i += 256){
        atomicAdd(&p.S->gsum[i], gS[i]);
        atomicAdd(&p.S->gsq[i],  gQ[i]);
    }
}

__global__ __launch_bounds__(256) void k_gstats(P p){
    int i = blockIdx.x*256 + threadIdx.x;
    if (i >= 1024) return;
    int g = i >> 7;
    float c = fmaxf(p.S->gcnt[g], 1.0f);
    float m = p.S->gsum[i] / c;
    float q = p.S->gsq[i]  / c;
    p.S->gmean[i] = m;
    p.S->gstd[i]  = sqrtf(fmaxf(q - m*m, 1e-8f));
}

// ---------------- gate GEMM: sigmoid([h2, sub_enh] @ gate_W) ; h_fused ----------------
__global__ __launch_bounds__(128) void k_gate(P p){
    __shared__ __align__(16) __bf16 At[4][16*264];
    int tid = threadIdx.x, wv = tid >> 5, ln = tid & 31;
    int base = blockIdx.x*64 + wv*16;
    bool fullt = (base + 16) <= N_NODES;
    {
        int m = ln >> 1, half = ln & 1;
        int n = base + m;
        unsigned short* arow = (unsigned short*)&At[wv][m*264];
        if (n < N_NODES){
            const float* hr = p.H + (size_t)n*128 + half*64;
            #pragma unroll
            for (int j = 0; j < 64; j += 4){
                float4 v = *(const float4*)(hr + j);
                arow[half*64+j+0] = f2bf(v.x);
                arow[half*64+j+1] = f2bf(v.y);
                arow[half*64+j+2] = f2bf(v.z);
                arow[half*64+j+3] = f2bf(v.w);
            }
            float sub[4];
            #pragma unroll
            for (int k = 0; k < 4; ++k) sub[k] = p.x[(size_t)n*XSTRIDE + 128 + k];
            for (int j = 0; j < 64; ++j){
                int c = half*64 + j;
                float pre = p.enh_b[c];
                #pragma unroll
                for (int k = 0; k < 4; ++k) pre += sub[k] * p.enh_W[k*128 + c];
                float se = fmaxf(p.S->aE[c]*pre + p.S->bE[c], 0.f);
                arow[128 + c] = f2bf(se);
            }
        } else {
            for (int j = 0; j < 64; ++j){
                arow[half*64 + j] = 0;
                arow[128 + half*64 + j] = 0;
            }
        }
    }
    const __bf16* Wg = (const __bf16*)p.wbG;
    const __bf16* arow = &At[wv][(ln & 15)*264];
    int half = ln >> 4;
    for (int nt = 0; nt < 8; ++nt){
        v8f acc = {0,0,0,0,0,0,0,0};
        #pragma unroll
        for (int kt = 0; kt < 8; ++kt){
            v16bf a = ld_a(arow, kt, half);
            v16bf b = ld_b(&Wg[(kt*32 + half*8)*128 + nt*16 + (ln & 15)], 128);
            acc = wmma_bf(a, b, acc);
        }
        int col = nt*16 + (ln & 15);
        float bias = p.gate_b[col];
        const unsigned short* tbl = (const unsigned short*)&At[wv][0];
        if (fullt){
            #pragma unroll
            for (int k = 0; k < 8; ++k){
                int r = k + 8*half;
                float gate = sigf(acc[k] + bias);
                float h2 = bf2f(tbl[r*264 + col]);
                float se = bf2f(tbl[r*264 + 128 + col]);
                p.hfus[(size_t)(base + r)*128 + col] = h2 + gate*se;
            }
        } else {
            #pragma unroll
            for (int k = 0; k < 8; ++k){
                int r = k + 8*half;
                int n = base + r;
                if (n >= N_NODES) continue;
                float gate = sigf(acc[k] + bias);
                float h2 = bf2f(tbl[r*264 + col]);
                float se = bf2f(tbl[r*264 + 128 + col]);
                p.hfus[(size_t)n*128 + col] = h2 + gate*se;
            }
        }
    }
}

// ---------------- score pass 1: [h_fused, h_dev] @ score_W1 + BN stats ----------------
__global__ __launch_bounds__(128) void k_score1(P p){
    __shared__ __align__(16) __bf16 At[4][16*264];
    __shared__ float sS[128], sQ[128];
    int tid = threadIdx.x, wv = tid >> 5, ln = tid & 31;
    if (tid < 128){ sS[tid] = 0.f; sQ[tid] = 0.f; }
    __syncthreads();
    int base = blockIdx.x*64 + wv*16;
    bool fullt = (base + 16) <= N_NODES;
    {
        int m = ln >> 1, half = ln & 1;
        int n = base + m;
        unsigned short* arow = (unsigned short*)&At[wv][m*264];
        if (n < N_NODES){
            const float* fr = p.hfus + (size_t)n*128 + half*64;
            #pragma unroll
            for (int j = 0; j < 64; j += 4){
                float4 v = *(const float4*)(fr + j);
                arow[half*64+j+0] = f2bf(v.x);
                arow[half*64+j+1] = f2bf(v.y);
                arow[half*64+j+2] = f2bf(v.z);
                arow[half*64+j+3] = f2bf(v.w);
            }
            int g = p.batch[n];
            const float* hr = p.H + (size_t)n*128 + half*64;
            #pragma unroll
            for (int j = 0; j < 64; j += 4){
                float4 v = *(const float4*)(hr + j);
                int c = half*64 + j;
                arow[128+c+0] = f2bf((v.x - p.S->gmean[g*128+c+0]) / (p.S->gstd[g*128+c+0] + 1e-8f));
                arow[128+c+1] = f2bf((v.y - p.S->gmean[g*128+c+1]) / (p.S->gstd[g*128+c+1] + 1e-8f));
                arow[128+c+2] = f2bf((v.z - p.S->gmean[g*128+c+2]) / (p.S->gstd[g*128+c+2] + 1e-8f));
                arow[128+c+3] = f2bf((v.w - p.S->gmean[g*128+c+3]) / (p.S->gstd[g*128+c+3] + 1e-8f));
            }
        } else {
            for (int j = 0; j < 64; ++j){
                arow[half*64 + j] = 0;
                arow[128 + half*64 + j] = 0;
            }
        }
    }
    const __bf16* Wg = (const __bf16*)p.wbS;
    const __bf16* arow = &At[wv][(ln & 15)*264];
    int half = ln >> 4;
    for (int nt = 0; nt < 8; ++nt){
        v8f acc = {0,0,0,0,0,0,0,0};
        #pragma unroll
        for (int kt = 0; kt < 8; ++kt){
            v16bf a = ld_a(arow, kt, half);
            v16bf b = ld_b(&Wg[(kt*32 + half*8)*128 + nt*16 + (ln & 15)], 128);
            acc = wmma_bf(a, b, acc);
        }
        int col = nt*16 + (ln & 15);
        float bias = p.score_b1[col];
        float ps = 0.f, pq = 0.f;
        if (fullt){
            #pragma unroll
            for (int k = 0; k < 8; ++k){
                int n = base + k + 8*half;
                float y = acc[k] + bias;
                p.y1[(size_t)n*128 + col] = y;
                ps += y; pq += y*y;
            }
        } else {
            #pragma unroll
            for (int k = 0; k < 8; ++k){
                int n = base + k + 8*half;
                float y = acc[k] + bias;
                if (n < N_NODES){
                    p.y1[(size_t)n*128 + col] = y;
                    ps += y; pq += y*y;
                }
            }
        }
        ps += __shfl_xor(ps, 16, 32);
        pq += __shfl_xor(pq, 16, 32);
        if (ln < 16){
            atomicAdd(&sS[col], ps);
            atomicAdd(&sQ[col], pq);
        }
    }
    __syncthreads();
    if (tid < 128){
        atomicAdd(&p.S->csS[tid],  sS[tid]);
        atomicAdd(&p.S->csqS[tid], sQ[tid]);
    }
}

__global__ __launch_bounds__(128) void k_coefS(P p){
    int c = threadIdx.x;
    if (c >= 128) return;
    float inv = 1.0f / (float)N_NODES;
    float m = p.S->csS[c]*inv;
    float v = p.S->csqS[c]*inv - m*m;
    float a = p.score_g[c] * rsqrtf(v + BN_EPS);
    p.S->aS[c] = a;
    p.S->bS[c] = p.score_beta[c] - m*a;
}

// ---------------- score pass 2: logit per node + segment max ----------------
__global__ __launch_bounds__(256) void k_score2(P p){
    int wv = threadIdx.x >> 5, ln = threadIdx.x & 31;
    int n = blockIdx.x*8 + wv;
    if (n >= N_NODES) return;
    float pl = 0.f;
    #pragma unroll
    for (int j = 0; j < 4; ++j){
        int c = ln + 32*j;
        float s = fmaxf(p.S->aS[c]*p.y1[(size_t)n*128 + c] + p.S->bS[c], 0.f);
        pl += s * p.score_W2[c];
    }
    #pragma unroll
    for (int o = 16; o > 0; o >>= 1) pl += __shfl_xor(pl, o, 32);
    if (ln == 0){
        float l = pl + p.score_b2[0];
        p.logit[n] = l;
        atomicMax(&p.S->smax_enc[p.batch[n]], encf(l));
    }
}

// ---------------- softmax pooling: denom + sum(e * h_fused) per graph ----------------
__global__ __launch_bounds__(256) void k_pool(P p){
    __shared__ float wS[1024];
    __shared__ float den[8];
    int tid = threadIdx.x, wv = tid >> 5, ln = tid & 31;
    for (int i = tid; i < 1024; i += 256) wS[i] = 0.f;
    if (tid < 8) den[tid] = 0.f;
    __syncthreads();
    for (int n = blockIdx.x*8 + wv; n < N_NODES; n += 256*8){
        int g = p.batch[n];
        float sm = decf(p.S->smax_enc[g]);
        if (!finitef(sm)) sm = 0.f;
        float e = expf(p.logit[n] - sm);
        if (ln == 0) atomicAdd(&den[g], e);
        #pragma unroll
        for (int j = 0; j < 4; ++j){
            int c = ln + 32*j;
            atomicAdd(&wS[g*128 + c], e * p.hfus[(size_t)n*128 + c]);
        }
    }
    __syncthreads();
    for (int i = tid; i < 1024; i += 256) atomicAdd(&p.S->wsum[i], wS[i]);
    if (tid < 8) atomicAdd(&p.S->denom[tid], den[tid]);
}

// ---------------- final: out = seg_max(z0) + pooled @ lin1 ----------------
__global__ __launch_bounds__(64) void k_final(P p){
    int t = threadIdx.x;
    if (t >= 16) return;
    int g = t >> 1, tt = t & 1;
    float alpha = sigf(p.mix[0]);
    float dn = fmaxf(p.S->denom[g], 1e-30f);
    float acc = p.lin1_b[tt];
    for (int c = 0; c < 128; ++c){
        float wf = p.S->wsum[g*128 + c] / dn;
        float wo = p.S->gmean[g*128 + c];
        acc += (alpha*wf + (1.f - alpha)*wo) * p.lin1_W[c*2 + tt];
    }
    float base = decf(p.S->out_enc[t]);
    if (!finitef(base)) base = 0.f;
    p.out[t] = base + acc;
}

// ---------------- host launcher ----------------
extern "C" void kernel_launch(void* const* d_in, const int* in_sizes, int n_in,
                              void* d_out, int out_size, void* d_ws, size_t ws_size,
                              hipStream_t stream){
    (void)in_sizes; (void)n_in; (void)out_size; (void)ws_size;
    P p;
    p.x        = (const float*)d_in[0];
    p.ei       = (const int*)  d_in[1];
    p.batch    = (const int*)  d_in[2];
    p.attn_W1  = (const float*)d_in[3];  p.attn_b1   = (const float*)d_in[4];
    p.attn_W2  = (const float*)d_in[5];  p.attn_b2   = (const float*)d_in[6];
    p.first_W  = (const float*)d_in[7];  p.first_b   = (const float*)d_in[8];
    p.first_g  = (const float*)d_in[9];  p.first_beta= (const float*)d_in[10];
    p.proj0_W  = (const float*)d_in[11]; p.proj0_b   = (const float*)d_in[12];
    p.proj1_W  = (const float*)d_in[13]; p.proj1_b   = (const float*)d_in[14];
    p.scale0   = (const float*)d_in[15]; p.scale1    = (const float*)d_in[16];
    p.lin0_W   = (const float*)d_in[17]; p.lin0_b    = (const float*)d_in[18];
    p.lin1_W   = (const float*)d_in[19]; p.lin1_b    = (const float*)d_in[20];
    p.conv_W   = (const float*)d_in[21]; p.conv_b    = (const float*)d_in[22];
    p.conv_g   = (const float*)d_in[23]; p.conv_beta = (const float*)d_in[24];
    p.enh_W    = (const float*)d_in[25]; p.enh_b     = (const float*)d_in[26];
    p.enh_g    = (const float*)d_in[27]; p.enh_beta  = (const float*)d_in[28];
    p.gate_W   = (const float*)d_in[29]; p.gate_b    = (const float*)d_in[30];
    p.score_W1 = (const float*)d_in[31]; p.score_b1  = (const float*)d_in[32];
    p.score_g  = (const float*)d_in[33]; p.score_beta= (const float*)d_in[34];
    p.score_W2 = (const float*)d_in[35]; p.score_b2  = (const float*)d_in[36];
    p.mix      = (const float*)d_in[37];

    char* w = (char*)d_ws;
    p.h1b   = (unsigned*)w;        w += (size_t)N_NODES*128*2;
    p.y1    = (float*)w;           w += (size_t)N_NODES*128*4;
    p.hfus  = (float*)w;           w += (size_t)N_NODES*128*4;
    p.logit = (float*)w;           w += (size_t)N_NODES*4;
    p.wbC   = (unsigned short*)w;  w += 256*128*2;
    p.wbG   = (unsigned short*)w;  w += 256*128*2;
    p.wbS   = (unsigned short*)w;  w += 256*128*2;
    char* zstart = w;
    p.hnew     = (unsigned*)w;     w += (size_t)N_NODES*128*4;
    p.attn_sum = (float*)w;        w += (size_t)N_NODES*4;
    p.degree   = (float*)w;        w += (size_t)N_NODES*4;
    p.S        = (Small*)w;        w += sizeof(Small);
    size_t zbytes = (size_t)(w - zstart);

    p.out = (float*)d_out;
    p.Z   = (float*)d_out + 16;
    p.H   = (float*)d_out + 16 + (size_t)N_NODES*2;

    hipMemsetAsync(zstart, 0, zbytes, stream);

    k_cvtW      <<<128,  256, 0, stream>>>(p);
    k_edgeattn  <<<3125, 256, 0, stream>>>(p);
    k_submom    <<<196,  256, 0, stream>>>(p);
    k_layer1    <<<782,  128, 0, stream>>>(p);
    k_coef1     <<<1,    128, 0, stream>>>(p);
    k_apply1    <<<6250, 256, 0, stream>>>(p);
    k_conv_stats<<<CONV_BLOCKS, 128, 0, stream>>>(p);
    k_coefC     <<<1,    128, 0, stream>>>(p);
    k_conv_apply<<<CONV_BLOCKS, 128, 0, stream>>>(p);
    k_node2     <<<NB2,  256, 0, stream>>>(p);
    k_gstats    <<<4,    256, 0, stream>>>(p);
    k_gate      <<<782,  128, 0, stream>>>(p);
    k_score1    <<<782,  128, 0, stream>>>(p);
    k_coefS     <<<1,    128, 0, stream>>>(p);
    k_score2    <<<6250, 256, 0, stream>>>(p);
    k_pool      <<<256,  256, 0, stream>>>(p);
    k_final     <<<1,    64,  0, stream>>>(p);
}
